// GraphNet_73246372266487
// MI455X (gfx1250) — matrix operand order repaired and checked by reading, not compiled, not run on hardware
//
#include <hip/hip_runtime.h>

// ---------------------------------------------------------------------------
// GraphNet for MI455X (gfx1250, wave32, WMMA bf16 16x16x32, f32 accumulate)
// ---------------------------------------------------------------------------
// Shapes: B=8, C=512, H=W=64, BNUM=4 (16 blocks of 16x16 pixels), BNOD=8,
// NODE_NUM=128, n=256 pixels/block.
// ---------------------------------------------------------------------------

typedef __attribute__((ext_vector_type(16))) __bf16 v16bf;
typedef __attribute__((ext_vector_type(8)))  __bf16 v8bf;
typedef __attribute__((ext_vector_type(8)))  float  v8f;

#define WMMA_BF16(A, B, C) \
  __builtin_amdgcn_wmma_f32_16x16x32_bf16(false, (A), false, (B), (short)0, (C), false, false)

__device__ __forceinline__ v16bf cat8(v8bf lo, v8bf hi) {
  return __builtin_shufflevector(lo, hi, 0,1,2,3,4,5,6,7,8,9,10,11,12,13,14,15);
}

// ------------------------------ workspace map ------------------------------
// [0        , 262144)  w2pack  bf16 : A-operand packed (k,kc,lane,i)
// [262144   , 524288)  awpack  bf16
// [524288   , 525312)  cnst    f32[128]
// [525312   , 2622464) nodes_pre f32[8][128][512]
// [2622464  , 2626560) rowsq   f32[8][128]
// [2626560  , 2630656) rowinv  f32[8][128]
// [2630656  , 2630912) ginv    f32[8]
// [2630912  , 3679488) gnodes  bf16[8][128][512]
// [3679488  , 4203776) adj_pre f32[8][128][128]
// [4203776  , 4205824) red     f32[512]
// [4205824  , 4205832) minmax  f32[2]
// ---------------------------------------------------------------------------

// K0: per (k,d) node: inv = 1/sigmoid(sigma) = 1+e^-s; pack w2=inv^2 and
// aw=a*inv^2 into the exact WMMA 16x32 bf16 A-operand lane/element layout:
// lane = half*16 + d (rows 8..15 zeroed); element i maps K-chunk offsets
// {0..7,16..23} (lanes<16) / {8..15,24..31} (lanes>=16). cnst[d]=sum a^2 inv^2.
__global__ void k0_prep(const float* __restrict__ anchor,
                        const float* __restrict__ sigma,
                        __bf16* __restrict__ w2pack,
                        __bf16* __restrict__ awpack,
                        float* __restrict__ cnst) {
  int kd = blockIdx.x;            // 0..127 -> (k, d)
  int k = kd >> 3, d = kd & 7;
  int t = threadIdx.x;            // 256
  __shared__ float red[256];
  float part = 0.f;
  for (int c = t; c < 512; c += 256) {
    float a   = anchor[(size_t)kd * 512 + c];
    float s   = sigma [(size_t)kd * 512 + c];
    float inv = 1.0f + __expf(-s);   // 1/sigmoid(s)
    float w2  = inv * inv;
    float aw  = a * w2;
    int kc   = c >> 5;
    int cc   = c & 31;
    int half = (cc >> 3) & 1;                     // K in {8..15,24..31} -> upper lanes
    int i    = (cc & 7) + ((cc & 16) ? 8 : 0);    // element index within lane
    size_t base = (((size_t)(k * 16 + kc)) * 32 + half * 16) * 16 + i;
    w2pack[base + (size_t)d * 16]       = (__bf16)w2;
    w2pack[base + (size_t)(d + 8) * 16] = (__bf16)0.f;   // zero rows 8..15
    awpack[base + (size_t)d * 16]       = (__bf16)aw;
    awpack[base + (size_t)(d + 8) * 16] = (__bf16)0.f;
    part += a * a * w2;
  }
  red[t] = part;
  __syncthreads();
  for (int s2 = 128; s2 > 0; s2 >>= 1) {
    if (t < s2) red[t] += red[t + s2];
    __syncthreads();
  }
  if (t == 0) cnst[kd] = red[0];
}

// K1: per (b, block k). 256 threads = 8 waves.
// Pass 1: logits^T[d,p] via two WMMA chains (A = packed w2/aw, B = X / X^2
//         staged in LDS as [p][c], c-contiguous). Per-lane softmax over d.
// Pass 2: S[d,c] = SA*X via WMMA (A = saT bf16 in LDS, B = X^T staged [c][p]),
//         then nodes_pre = inv*(S - a*den)/(den+1e-9).
__global__ void __launch_bounds__(256)
k1_main(const float* __restrict__ x,
        const float* __restrict__ anchor,
        const float* __restrict__ sigma,
        const __bf16* __restrict__ w2pack,
        const __bf16* __restrict__ awpack,
        const float* __restrict__ cnst,
        float* __restrict__ nodes_pre,
        float* __restrict__ out_sa) {
  const int bid = blockIdx.x;            // 0..127 = b*16 + k
  const int b   = bid >> 4;
  const int k   = bid & 15;
  const int bi  = k >> 2, bj = k & 3;    // block row/col in 4x4 grid
  const int t    = threadIdx.x;
  const int wave = t >> 5;
  const int lane = t & 31;
  const int l16  = lane & 15;
  const int lh   = lane >> 4;            // lane half (0/1)

  // pitch 264 bf16 = 528 B = 33*16 B (16-B aligned rows, 4-word bank skew)
  __shared__ __attribute__((aligned(16))) __bf16 Xs[256][264];   // 135168 B
  __shared__ __attribute__((aligned(16))) __bf16 saT[16][264];   //   8448 B
  __shared__ float denS[8];
  __shared__ float cnstS[8];

  if (t < 8) { cnstS[t] = cnst[k * 8 + t]; denS[t] = 0.f; }
  for (int idx = t; idx < 16 * 264; idx += 256)
    saT[idx / 264][idx % 264] = (__bf16)0.f;

  const size_t xbase = (size_t)b * 512 * 4096;
  const int pr = t >> 4, pc = t & 15;            // this thread's pixel (row-major)
  const int prow0 = bi * 16, pcol0 = bj * 16;

  v8f accQ[2] = {};   // sum x^2 * w2  (logits tiles: M=d, N=p)
  v8f accL[2] = {};   // sum x   * aw

  // ----------------------------- pass 1 -----------------------------------
  for (int hf = 0; hf < 2; ++hf) {
    __syncthreads();
    // stage X half into LDS as [p][c_local] (coalesced global reads)
    for (int cl = 0; cl < 256; ++cl) {
      int c = hf * 256 + cl;
      float xv = x[xbase + (size_t)c * 4096 + (size_t)(prow0 + pr) * 64 + (pcol0 + pc)];
      Xs[t][cl] = (__bf16)xv;
    }
    __syncthreads();
    for (int kc2 = 0; kc2 < 8; ++kc2) {          // 8 K-chunks of 32 channels
      int kc = hf * 8 + kc2;
      const v16bf Aq = *(const v16bf*)(w2pack + (((size_t)(k * 16 + kc)) * 32 + lane) * 16);
      const v16bf Al = *(const v16bf*)(awpack + (((size_t)(k * 16 + kc)) * 32 + lane) * 16);
      int cb = kc2 * 32 + lh * 8;                // B: lanes<16 K={0..7,16..23}, lanes>=16 K={8..15,24..31}
      #pragma unroll
      for (int tt = 0; tt < 2; ++tt) {           // each wave owns 2 pixel tiles
        int prow = (wave * 2 + tt) * 16 + l16;   // B column = pixel
        v8bf blo = *(const v8bf*)&Xs[prow][cb];
        v8bf bhi = *(const v8bf*)&Xs[prow][cb + 16];
        v16bf Bx = cat8(blo, bhi);
        v16bf Bx2;
        #pragma unroll
        for (int e = 0; e < 16; ++e) {
          float f = (float)Bx[e];
          Bx2[e] = (__bf16)(f * f);
        }
        accQ[tt] = WMMA_BF16(Aq, Bx2, accQ[tt]);
        accL[tt] = WMMA_BF16(Al, Bx,  accL[tt]);
      }
    }
  }
  __syncthreads();

  // softmax over the 8 nodes: lanes 0..15 hold rows d=0..7 in acc[0..7]
  if (lh == 0) {
    #pragma unroll
    for (int tt = 0; tt < 2; ++tt) {
      int p = (wave * 2 + tt) * 16 + l16;
      float lg[8];
      float m = -3.4e38f;
      #pragma unroll
      for (int r = 0; r < 8; ++r) {
        lg[r] = -0.5f * accQ[tt][r] + accL[tt][r] - 0.5f * cnstS[r];
        m = fmaxf(m, lg[r]);
      }
      float ev[8], ssum = 0.f;
      #pragma unroll
      for (int r = 0; r < 8; ++r) { ev[r] = __expf(lg[r] - m); ssum += ev[r]; }
      float invs = 1.f / ssum;
      #pragma unroll
      for (int r = 0; r < 8; ++r) {
        float sa = ev[r] * invs;
        saT[r][p] = (__bf16)sa;                                  // A-operand for pass 2
        out_sa[((size_t)b * 128 + k * 8 + r) * 256 + p] = sa;    // soft_assign output
        atomicAdd(&denS[r], sa);                                 // ds_add_f32
      }
    }
  }

  // ----------------------------- pass 2 -----------------------------------
  for (int hf = 0; hf < 2; ++hf) {
    __syncthreads();
    // stage X half transposed: Xs[c_local][p] (contiguous LDS writes over p)
    for (int cl = 0; cl < 256; ++cl) {
      int c = hf * 256 + cl;
      float xv = x[xbase + (size_t)c * 4096 + (size_t)(prow0 + pr) * 64 + (pcol0 + pc)];
      Xs[cl][t] = (__bf16)xv;
    }
    __syncthreads();
    v8f accS[2] = {};
    for (int pcn = 0; pcn < 8; ++pcn) {          // 8 K-chunks of 32 pixels
      int p0 = pcn * 32;
      int kb = lh * 8;                            // A element->K mapping
      v8bf alo = *(const v8bf*)&saT[l16][p0 + kb];
      v8bf ahi = *(const v8bf*)&saT[l16][p0 + 16 + kb];
      v16bf Asa = cat8(alo, ahi);
      #pragma unroll
      for (int tt = 0; tt < 2; ++tt) {           // each wave owns 2 channel tiles
        int cl = (wave * 2 + tt) * 16 + l16;     // B column = channel
        v8bf blo = *(const v8bf*)&Xs[cl][p0 + lh * 8];
        v8bf bhi = *(const v8bf*)&Xs[cl][p0 + lh * 8 + 16];
        v16bf Bx = cat8(blo, bhi);
        accS[tt] = WMMA_BF16(Asa, Bx, accS[tt]);
      }
    }
    if (lh == 0) {                               // rows d=0..7 live in lanes 0..15
      #pragma unroll
      for (int tt = 0; tt < 2; ++tt) {
        int cg = hf * 256 + (wave * 2 + tt) * 16 + l16;
        #pragma unroll
        for (int r = 0; r < 8; ++r) {
          int node = k * 8 + r;
          float a   = anchor[(size_t)node * 512 + cg];
          float s   = sigma [(size_t)node * 512 + cg];
          float inv = 1.f + __expf(-s);
          float den = denS[r];
          float num = inv * (accS[tt][r] - a * den);
          nodes_pre[((size_t)b * 128 + node) * 512 + cg] = num / (den + 1e-9f);
        }
      }
    }
  }
}

// K2: per (b,node) row: sum of squares -> rowsq, rowinv = 1/max(norm,1e-12)
__global__ void k2_rowsq(const float* __restrict__ nodes_pre,
                         float* __restrict__ rowsq,
                         float* __restrict__ rowinv) {
  int bn = blockIdx.x;       // 0..1023
  int t  = threadIdx.x;      // 256
  __shared__ float red[256];
  float v0 = nodes_pre[(size_t)bn * 512 + t];
  float v1 = nodes_pre[(size_t)bn * 512 + t + 256];
  red[t] = v0 * v0 + v1 * v1;
  __syncthreads();
  for (int s = 128; s > 0; s >>= 1) {
    if (t < s) red[t] += red[t + s];
    __syncthreads();
  }
  if (t == 0) {
    float ss = red[0];
    rowsq[bn]  = ss;
    rowinv[bn] = 1.f / fmaxf(sqrtf(ss), 1e-12f);
  }
}

// K3: per batch: global norm of the row-normalized flat vector
__global__ void k3_ginv(const float* __restrict__ rowsq,
                        const float* __restrict__ rowinv,
                        float* __restrict__ ginv) {
  int b = blockIdx.x;
  int t = threadIdx.x;       // 128
  __shared__ float red[128];
  float ri = rowinv[b * 128 + t];
  red[t] = rowsq[b * 128 + t] * ri * ri;
  __syncthreads();
  for (int s = 64; s > 0; s >>= 1) {
    if (t < s) red[t] += red[t + s];
    __syncthreads();
  }
  if (t == 0) ginv[b] = 1.f / fmaxf(sqrtf(red[0]), 1e-12f);
}

// K4: final normalize; write graph output [B,C,node] and bf16 copy for adj GEMM
__global__ void k4_norm(const float* __restrict__ nodes_pre,
                        const float* __restrict__ rowinv,
                        const float* __restrict__ ginv,
                        float* __restrict__ graph_out,
                        __bf16* __restrict__ gnodes) {
  int bn = blockIdx.x;               // b*128 + node
  int b = bn >> 7, node = bn & 127;
  int t = threadIdx.x;
  float sc = rowinv[bn] * ginv[b];
  for (int c = t; c < 512; c += 256) {
    float g = nodes_pre[(size_t)bn * 512 + c] * sc;
    graph_out[((size_t)b * 512 + c) * 128 + node] = g;
    gnodes[(size_t)bn * 512 + c] = (__bf16)g;
  }
}

// K5: adj = G * G^T + posbias  via WMMA, one 16x16 tile per wave, K=512
__global__ void __launch_bounds__(256)
k5_adj(const __bf16* __restrict__ gnodes,
       const float* __restrict__ pos_table,
       float* __restrict__ adj_pre) {
  int t = threadIdx.x;
  int wave = t >> 5, lane = t & 31, l16 = lane & 15, lh = lane >> 4;
  int gw = blockIdx.x * 8 + wave;    // 0..511 = b*64 + tile
  int b = gw >> 6;
  int tile = gw & 63;
  int nt = tile >> 3, mt = tile & 7;
  const __bf16* gb = gnodes + (size_t)b * 128 * 512;
  v8f acc = {};
  for (int kc = 0; kc < 16; ++kc) {
    int kb = lh * 16;
    const __bf16* arow = gb + (size_t)(nt * 16 + l16) * 512 + kc * 32;
    v8bf alo = *(const v8bf*)(arow + kb);
    v8bf ahi = *(const v8bf*)(arow + kb + 8);
    v16bf A = cat8(alo, ahi);
    v16bf B = *(const v16bf*)(gb + (size_t)(mt * 16 + l16) * 512 + kc * 32 + lh * 16);
    acc = WMMA_BF16(A, B, acc);
  }
  #pragma unroll
  for (int r = 0; r < 8; ++r) {
    int nrow = nt * 16 + (lh ? r + 8 : r);
    int mcol = mt * 16 + l16;
    int kbn = nrow >> 3, kbm = mcol >> 3;         // block indices (Swin rel pos)
    int iy = (kbn >> 2) - (kbm >> 2) + 3;
    int ix = (kbn & 3)  - (kbm & 3)  + 3;
    float bias = pos_table[iy * 7 + ix];
    adj_pre[((size_t)b * 128 + nrow) * 128 + mcol] = acc[r] + bias;
  }
}

// K6/K7: global min/max of adj (two-stage reduction)
__global__ void k6_minmax(const float* __restrict__ adj_pre, float* __restrict__ red) {
  int t = threadIdx.x;
  size_t i = (size_t)blockIdx.x * 512 + t;
  float v0 = adj_pre[i], v1 = adj_pre[i + 256];
  __shared__ float smin[256], smax[256];
  smin[t] = fminf(v0, v1);
  smax[t] = fmaxf(v0, v1);
  __syncthreads();
  for (int s = 128; s > 0; s >>= 1) {
    if (t < s) { smin[t] = fminf(smin[t], smin[t + s]); smax[t] = fmaxf(smax[t], smax[t + s]); }
    __syncthreads();
  }
  if (t == 0) { red[blockIdx.x] = smin[0]; red[256 + blockIdx.x] = smax[0]; }
}

__global__ void k7_final(const float* __restrict__ red, float* __restrict__ mm) {
  int t = threadIdx.x;
  __shared__ float smin[256], smax[256];
  smin[t] = red[t];
  smax[t] = red[256 + t];
  __syncthreads();
  for (int s = 128; s > 0; s >>= 1) {
    if (t < s) { smin[t] = fminf(smin[t], smin[t + s]); smax[t] = fmaxf(smax[t], smax[t + s]); }
    __syncthreads();
  }
  if (t == 0) { mm[0] = smin[0]; mm[1] = smax[0]; }
}

// K8: min-max normalize adj into output
__global__ void k8_normadj(const float* __restrict__ adj_pre,
                           const float* __restrict__ mm,
                           float* __restrict__ adj_out) {
  size_t i = (size_t)blockIdx.x * 256 + threadIdx.x;
  float amin = mm[0], amax = mm[1];
  adj_out[i] = (adj_pre[i] - amin) / (1e-4f + (amax - amin));
}

// ---------------------------------------------------------------------------
extern "C" void kernel_launch(void* const* d_in, const int* in_sizes, int n_in,
                              void* d_out, int out_size, void* d_ws, size_t ws_size,
                              hipStream_t stream) {
  (void)in_sizes; (void)n_in; (void)out_size; (void)ws_size;
  const float* x         = (const float*)d_in[0];
  const float* anchor    = (const float*)d_in[1];
  const float* sigma     = (const float*)d_in[2];
  const float* pos_table = (const float*)d_in[3];

  float* out       = (float*)d_out;
  float* out_graph = out;                       // [8,512,128]
  float* out_adj   = out + 8 * 512 * 128;       // [8,128,128]
  float* out_sa    = out_adj + 8 * 128 * 128;   // [8,128,256]

  char* ws = (char*)d_ws;
  __bf16* w2pack    = (__bf16*)(ws + 0);
  __bf16* awpack    = (__bf16*)(ws + 262144);
  float*  cnst      = (float*) (ws + 524288);
  float*  nodes_pre = (float*) (ws + 525312);
  float*  rowsq     = (float*) (ws + 2622464);
  float*  rowinv    = (float*) (ws + 2626560);
  float*  ginv      = (float*) (ws + 2630656);
  __bf16* gnodes    = (__bf16*)(ws + 2630912);
  float*  adj_pre   = (float*) (ws + 3679488);
  float*  red       = (float*) (ws + 4203776);
  float*  mm        = (float*) (ws + 4205824);

  k0_prep   <<<128,  256, 0, stream>>>(anchor, sigma, w2pack, awpack, cnst);
  k1_main   <<<128,  256, 0, stream>>>(x, anchor, sigma, w2pack, awpack, cnst,
                                       nodes_pre, out_sa);
  k2_rowsq  <<<1024, 256, 0, stream>>>(nodes_pre, rowsq, rowinv);
  k3_ginv   <<<8,    128, 0, stream>>>(rowsq, rowinv, ginv);
  k4_norm   <<<1024, 256, 0, stream>>>(nodes_pre, rowinv, ginv, out_graph, gnodes);
  k5_adj    <<<64,   256, 0, stream>>>(gnodes, pos_table, adj_pre);
  k6_minmax <<<256,  256, 0, stream>>>(adj_pre, red);
  k7_final  <<<1,    256, 0, stream>>>(red, mm);
  k8_normadj<<<512,  256, 0, stream>>>(adj_pre, mm, out_adj);
}
